// HierarchicalPathNetworkLayer_8564164788979
// MI455X (gfx1250) — matrix operand order repaired and checked by reading, not compiled, not run on hardware
//
#include <hip/hip_runtime.h>

// ---------------------------------------------------------------------------
// HierarchicalPathNetworkLayer for MI455X (gfx1250, wave32, WMMA)
//
// Upward:   h[i] = silu(seg0 @ Wup_top + seg1 @ Wup_bot + b)
// Downward: h[i-1] = silu(h[i-1] @ Wdn_top + hd @ Wdn_bot + b)
// GEMMs use v_wmma_f32_16x16x32_bf16 (bf16 inputs, f32 accumulate).
// Weights live in LDS pre-swizzled into per-lane fragment order; B fragments
// are single 32-byte contiguous LDS reads, pipelined one tile ahead.
// ---------------------------------------------------------------------------

#define FDIM 128

typedef __attribute__((ext_vector_type(16))) __bf16 v16bf;
typedef __attribute__((ext_vector_type(8)))  float  v8f;

// ---------------------------------------------------------------------------
// Zero-fill (graph-capture-safe alternative to hipMemsetAsync)
// ---------------------------------------------------------------------------
__global__ void hpn_zero_kernel(float4* __restrict__ p, long long n4) {
  long long i = (long long)blockIdx.x * blockDim.x + threadIdx.x;
  if (i < n4) p[i] = make_float4(0.f, 0.f, 0.f, 0.f);
}

// ---------------------------------------------------------------------------
// Edge scatter-add: one wave per edge, lane l moves float4 at cols [4l,4l+4).
// out[dst[e], :] += feat[src[e], :]
// ---------------------------------------------------------------------------
__global__ void hpn_scatter_add_kernel(const float* __restrict__ feat,
                                       const int* __restrict__ src,
                                       const int* __restrict__ dst,
                                       float* __restrict__ out, int E) {
  long long tid = (long long)blockIdx.x * blockDim.x + threadIdx.x;
  int e    = (int)(tid >> 5);
  int lane = threadIdx.x & 31;
  if (e >= E) return;
  size_t s = (size_t)src[e];
  size_t d = (size_t)dst[e];
  float4 v = ((const float4*)(feat + s * FDIM))[lane];
  float* dp = out + d * FDIM + (size_t)lane * 4;
  atomicAdd(dp + 0, v.x);
  atomicAdd(dp + 1, v.y);
  atomicAdd(dp + 2, v.z);
  atomicAdd(dp + 3, v.w);
}

// ---------------------------------------------------------------------------
// Fused split-GEMM + bias + SiLU helpers
// ---------------------------------------------------------------------------
static __device__ __forceinline__ float fsel(float4 v, int dn) {
  return dn == 0 ? v.x : dn == 1 ? v.y : dn == 2 ? v.z : v.w;
}

// Stage one 128x128 f32 weight half into LDS, bf16, fragment-swizzled:
//   slot s = (ks*8 + ct)*32 + lane holds 16 contiguous bf16:
//     col = ct*16 + (lane&15),  K(e) = ks*32 + e + 16*(lane>>4)
// Blocked 4x4 (k x n): 16 coalesced b128 global reads + 16 b64 LDS stores
// per thread (4 consecutive K elements are contiguous in the swizzle).
static __device__ __forceinline__ void stage_weights(
    const float* __restrict__ Wh, __bf16* Wlds, int tid) {
  const float4* Wh4 = (const float4*)Wh;
#pragma unroll
  for (int i = 0; i < 4; ++i) {
    int gid = i * 256 + tid;            // 0..1023: 32x32 grid of 4x4 blocks
    int kb = gid >> 5, nb = gid & 31;
    int k0 = kb * 4, n0 = nb * 4;
    float4 row[4];
#pragma unroll
    for (int dk = 0; dk < 4; ++dk) row[dk] = Wh4[(k0 + dk) * 32 + nb];
    int ks = k0 >> 5, hi2 = (k0 >> 4) & 1, e0 = k0 & 15;
#pragma unroll
    for (int dn = 0; dn < 4; ++dn) {
      int n = n0 + dn;
      int ct = n >> 4, ln2 = n & 15;
      int slot = (ks * 8 + ct) * 32 + hi2 * 16 + ln2;
      union { __bf16 h[4]; unsigned long long u; } p;
      p.h[0] = (__bf16)fsel(row[0], dn);
      p.h[1] = (__bf16)fsel(row[1], dn);
      p.h[2] = (__bf16)fsel(row[2], dn);
      p.h[3] = (__bf16)fsel(row[3], dn);
      *(unsigned long long*)&Wlds[slot * 16 + e0] = p.u;  // aligned 8B store
    }
  }
}

static __device__ __forceinline__ v16bf load_bfrag(const __bf16* Wlds, int ks,
                                                   int ct, int lane) {
  return *(const v16bf*)(Wlds + (size_t)((ks * 8 + ct) * 32 + lane) * 16);
}

// Accumulate A[M x 128] @ Wlds (one 128x128 half) into acc[8].
// A pointer is used directly (kernel arg) so loads stay global_load_b128.
static __device__ __forceinline__ void gemm_accum_half(
    const float* __restrict__ A, const __bf16* Wlds, int r, int lane, int hi,
    v8f acc[8]) {
  const float4* Arow4 = (const float4*)(A + (size_t)r * FDIM);
#pragma unroll
  for (int ks = 0; ks < 4; ++ks) {
    // A fragment (matches B swizzle): element e -> K = e + 16*hi,
    // i.e. one contiguous 16-float run.
    float4 q0 = Arow4[ks * 8 + 4 * hi];
    float4 q1 = Arow4[ks * 8 + 4 * hi + 1];
    float4 q2 = Arow4[ks * 8 + 4 * hi + 2];
    float4 q3 = Arow4[ks * 8 + 4 * hi + 3];
    v16bf a;
    a[0]  = (__bf16)q0.x; a[1]  = (__bf16)q0.y;
    a[2]  = (__bf16)q0.z; a[3]  = (__bf16)q0.w;
    a[4]  = (__bf16)q1.x; a[5]  = (__bf16)q1.y;
    a[6]  = (__bf16)q1.z; a[7]  = (__bf16)q1.w;
    a[8]  = (__bf16)q2.x; a[9]  = (__bf16)q2.y;
    a[10] = (__bf16)q2.z; a[11] = (__bf16)q2.w;
    a[12] = (__bf16)q3.x; a[13] = (__bf16)q3.y;
    a[14] = (__bf16)q3.z; a[15] = (__bf16)q3.w;
    // Software-pipeline B-fragment LDS loads one column tile ahead so the
    // wmma's operand load was issued an iteration earlier (no tight wait).
    v16bf bcur = load_bfrag(Wlds, ks, 0, lane);
#pragma unroll
    for (int ct = 0; ct < 8; ++ct) {
      v16bf bnext = (ct < 7) ? load_bfrag(Wlds, ks, ct + 1, lane) : bcur;
      acc[ct] = __builtin_amdgcn_wmma_f32_16x16x32_bf16(
          false, a, false, bcur, (short)0, acc[ct], false, false);
      bcur = bnext;
    }
  }
}

// ---------------------------------------------------------------------------
//   out[M x 128] = silu(A0 @ W[0:128,:] + A1 @ W[128:256,:] + bias)
// Block: 256 threads = 8 waves; block tile 128 rows x 128 cols.
// Wave w owns rows [blk*128 + 16w, +16): 8 accumulator tiles of 16x16.
// ---------------------------------------------------------------------------
__global__ __launch_bounds__(256) void hpn_gemm_silu_kernel(
    const float* __restrict__ A0, const float* __restrict__ A1,
    const float* __restrict__ W,        // [256 x 128] row-major f32
    const float* __restrict__ bias,     // [128]
    float* __restrict__ out, int M) {
  __shared__ __align__(32) __bf16 Wlds[128 * 128];  // 32 KB, fragment order

  const int wave = threadIdx.x >> 5;    // 0..7
  const int lane = threadIdx.x & 31;
  const int hi   = lane >> 4;           // lane half (0: lanes 0-15, 1: 16-31)
  const int ln   = lane & 15;
  const int row_base = blockIdx.x * 128 + wave * 16;

  v8f acc[8];
#pragma unroll
  for (int i = 0; i < 8; ++i)
#pragma unroll
    for (int v = 0; v < 8; ++v) acc[i][v] = 0.f;

  // Clamp tail rows (results for r>=M are computed but never stored)
  int r = row_base + ln;
  if (r >= M) r = M - 1;

  // ---- half 0: A0 @ W_top ----
  stage_weights(W, Wlds, threadIdx.x);
  __syncthreads();
  gemm_accum_half(A0, Wlds, r, lane, hi, acc);
  __syncthreads();                      // all waves done reading half 0
  // ---- half 1: A1 @ W_bot ----
  stage_weights(W + 128 * 128, Wlds, threadIdx.x);
  __syncthreads();
  gemm_accum_half(A1, Wlds, r, lane, hi, acc);

  // Epilogue: C/D layout (VGPR v, lane) -> row = v + 8*hi, col = lane&15
#pragma unroll
  for (int ct = 0; ct < 8; ++ct) {
    int c = ct * 16 + ln;
    float bv = bias[c];
#pragma unroll
    for (int v = 0; v < 8; ++v) {
      int rr = row_base + v + hi * 8;
      if (rr < M) {
        float x = acc[ct][v] + bv;
        out[(size_t)rr * FDIM + c] = x / (1.f + __expf(-x));  // SiLU
      }
    }
  }
}

// ---------------------------------------------------------------------------
// Host orchestration
// ---------------------------------------------------------------------------
extern "C" void kernel_launch(void* const* d_in, const int* in_sizes, int n_in,
                              void* d_out, int out_size, void* d_ws, size_t ws_size,
                              hipStream_t stream) {
  (void)in_sizes; (void)n_in; (void)out_size; (void)ws_size;
  const int N1 = 100000, N2 = 200000, N3 = 300000, N4 = 400000;

  const float* feat = (const float*)d_in[0];
  const float* w_up2 = (const float*)d_in[1];  const float* b_up2 = (const float*)d_in[2];
  const float* w_up3 = (const float*)d_in[3];  const float* b_up3 = (const float*)d_in[4];
  const float* w_up4 = (const float*)d_in[5];  const float* b_up4 = (const float*)d_in[6];
  const float* w_dn1 = (const float*)d_in[7];  const float* b_dn1 = (const float*)d_in[8];
  const float* w_dn2 = (const float*)d_in[9];  const float* b_dn2 = (const float*)d_in[10];
  const float* w_dn3 = (const float*)d_in[11]; const float* b_dn3 = (const float*)d_in[12];
  // Edge index arrays: for idx in (2,3,4), p in (0,1):
  //   base = 13 + (idx-2)*8 + p*4 : up_src, up_dst, down_src, down_dst
  const int* up_src[3][2];  const int* up_dst[3][2];
  const int* dn_src[3][2];  const int* dn_dst[3][2];
  for (int i = 0; i < 3; ++i)
    for (int p = 0; p < 2; ++p) {
      int base = 13 + i * 8 + p * 4;
      up_src[i][p] = (const int*)d_in[base + 0];
      up_dst[i][p] = (const int*)d_in[base + 1];
      dn_src[i][p] = (const int*)d_in[base + 2];
      dn_dst[i][p] = (const int*)d_in[base + 3];
    }

  // Workspace layout (floats): h2 | h3 | h4 | t0 | t1   (~870 MB)
  float* h2 = (float*)d_ws;
  float* h3 = h2 + (size_t)N2 * FDIM;
  float* h4 = h3 + (size_t)N3 * FDIM;
  float* t0 = h4 + (size_t)N4 * FDIM;
  float* t1 = t0 + (size_t)N4 * FDIM;

  auto zero = [&](float* p, long long nfloats) {
    long long n4 = nfloats / 4;
    int blocks = (int)((n4 + 255) / 256);
    hipLaunchKernelGGL(hpn_zero_kernel, dim3(blocks), dim3(256), 0, stream,
                       (float4*)p, n4);
  };
  auto scatter = [&](const float* f, const int* s, const int* d, float* o, int E) {
    int blocks = (E + 7) / 8;  // 8 waves (edges) per 256-thread block
    hipLaunchKernelGGL(hpn_scatter_add_kernel, dim3(blocks), dim3(256), 0, stream,
                       f, s, d, o, E);
  };
  auto gemm = [&](const float* a0, const float* a1, const float* w,
                  const float* b, float* o, int M) {
    int blocks = (M + 127) / 128;
    hipLaunchKernelGGL(hpn_gemm_silu_kernel, dim3(blocks), dim3(256), 0, stream,
                       a0, a1, w, b, o, M);
  };

  // ---- Upward pass ----
  // level 2
  zero(t0, (long long)N2 * FDIM); zero(t1, (long long)N2 * FDIM);
  scatter(feat, up_src[0][0], up_dst[0][0], t0, N2);
  scatter(feat, up_src[0][1], up_dst[0][1], t1, N2);
  gemm(t0, t1, w_up2, b_up2, h2, N2);
  // level 3
  zero(t0, (long long)N3 * FDIM); zero(t1, (long long)N3 * FDIM);
  scatter(h2, up_src[1][0], up_dst[1][0], t0, N3);
  scatter(h2, up_src[1][1], up_dst[1][1], t1, N3);
  gemm(t0, t1, w_up3, b_up3, h3, N3);
  // level 4
  zero(t0, (long long)N4 * FDIM); zero(t1, (long long)N4 * FDIM);
  scatter(h3, up_src[2][0], up_dst[2][0], t0, N4);
  scatter(h3, up_src[2][1], up_dst[2][1], t1, N4);
  gemm(t0, t1, w_up4, b_up4, h4, N4);

  // ---- Downward pass ----
  // idx=4 -> updates h3 (in place; blocks touch only their own 128-row window)
  zero(t0, (long long)N3 * FDIM);
  scatter(h4, dn_src[2][0], dn_dst[2][0], t0, N4);
  scatter(h4, dn_src[2][1], dn_dst[2][1], t0, N4);
  gemm(h3, t0, w_dn3, b_dn3, h3, N3);
  // idx=3 -> updates h2
  zero(t0, (long long)N2 * FDIM);
  scatter(h3, dn_src[1][0], dn_dst[1][0], t0, N3);
  scatter(h3, dn_src[1][1], dn_dst[1][1], t0, N3);
  gemm(h2, t0, w_dn2, b_dn2, h2, N2);
  // idx=2 -> produces h1 = output (feat is read-only input)
  zero(t0, (long long)N1 * FDIM);
  scatter(h2, dn_src[0][0], dn_dst[0][0], t0, N2);
  scatter(h2, dn_src[0][1], dn_dst[0][1], t0, N2);
  gemm(feat, t0, w_dn1, b_dn1, (float*)d_out, N1);
}
